// GNN_30039001268233
// MI455X (gfx1250) — matrix operand order repaired and checked by reading, not compiled, not run on hardware
//
#include <hip/hip_runtime.h>
#include <hip/hip_bf16.h>
#include <stdint.h>

// ---------------------------------------------------------------------------
// GCN 2-layer forward for MI455X (gfx1250, wave32).
//  - GEMMs via v_wmma_f32_16x16x32_bf16 (f32 accumulate), fully unrolled
//    with immediate-offset B-fragment loads from a pre-swizzled weight buffer.
//  - Edge message-passing via native f32 global atomics (L2-resident).
//  - Accumulators pre-initialized with self-loop + bias term to minimize
//    L2 traffic (no zero pass, no second epilogue).
// ---------------------------------------------------------------------------

typedef __attribute__((ext_vector_type(16))) __bf16 v16bf;
typedef __attribute__((ext_vector_type(8)))  __bf16 v8bf;
typedef __attribute__((ext_vector_type(8)))  float  v8f;

__device__ __forceinline__ void atomic_add_f32(float* p, float v) {
  // Relaxed agent-scope fp atomic -> global_atomic_add_f32 (no CAS loop).
  __hip_atomic_fetch_add(p, v, __ATOMIC_RELAXED, __HIP_MEMORY_SCOPE_AGENT);
}

// ---------------------------------------------------------------- utilities
__global__ void zero_kernel(uint32_t* __restrict__ p, long n) {
  long i = (long)blockIdx.x * blockDim.x + threadIdx.x;
  long stride = (long)gridDim.x * blockDim.x;
  for (; i < n; i += stride) p[i] = 0u;
}

__global__ void deg_kernel(const int* __restrict__ dst, int nE,
                           uint32_t* __restrict__ cnt) {
  int i = blockIdx.x * blockDim.x + threadIdx.x;
  if (i < nE) atomicAdd(&cnt[dst[i]], 1u);
}

// deg includes the self loop (+1); deg >= 1 always, so plain rsqrt.
__global__ void dinv_kernel(const uint32_t* __restrict__ cnt,
                            float* __restrict__ dinv, int n) {
  int i = blockIdx.x * blockDim.x + threadIdx.x;
  if (i < n) dinv[i] = rsqrtf((float)(cnt[i] + 1u));
}

__global__ void cvt_bf16_kernel(const float* __restrict__ x,
                                __bf16* __restrict__ xb, long n) {
  long i = (long)blockIdx.x * blockDim.x + threadIdx.x;
  if (i < n) xb[i] = (__bf16)x[i];
}

// Pre-swizzle W[k][n] (row-major KxN, f32) into bf16 WMMA-B fragment order:
// Wp[(((nb*4 + kb)*32 + lane)*16 + j] = W[kb*32 + 8*(lane>>4) + (j&7) + 16*(j>>3)][nb*16 + (lane&15)]
// -> each lane's 16-element B fragment is one contiguous 32B chunk, and all
//    fragments sit at immediate byte offsets (nb*4+kb)*1024 from lane*32.
__global__ void prep_w_kernel(const float* __restrict__ W,
                              __bf16* __restrict__ Wp, int ncols, int total) {
  int t = blockIdx.x * blockDim.x + threadIdx.x;
  if (t >= total) return;
  int j  = t & 15;
  int L  = (t >> 4) & 31;
  int kb = (t >> 9) & 3;
  int nb = t >> 11;
  int k = kb * 32 + ((L >> 4) << 3) + (j & 7) + ((j >> 3) << 4);
  int n = (nb << 4) + (L & 15);
  Wp[t] = (__bf16)W[k * ncols + n];
}

// ------------------------------------------------------------------- GEMM
// C[rows x NBLK*16] = A[rows x 128](bf16) * W[128 x NBLK*16], f32 accumulate.
// One wave per 16 rows; A-fragments hoisted and reused across all N-tiles.
// Fully unrolled: all B loads are immediate offsets from one base pointer.
template <int NBLK>
__global__ void __launch_bounds__(256)
gemm_bf16_kernel(const __bf16* __restrict__ A, const __bf16* __restrict__ Wp,
                 float* __restrict__ C, int rows) {
  constexpr int NCOLS = NBLK * 16;
  int lane = threadIdx.x & 31;
  int wave = (int)(((long)blockIdx.x * blockDim.x + threadIdx.x) >> 5);
  int m0 = wave << 4;
  if (m0 >= rows) return;              // wave-uniform: EXEC all-1s at WMMA
  int r    = lane & 15;
  int half = lane >> 4;

  // A fragment (16x32 bf16): lane holds M=r; K = half*8 + {j, j+16} chunks.
  const __bf16* abase = A + (long)(m0 + r) * 128 + half * 8;
  v16bf afr[4];
#pragma unroll
  for (int kb = 0; kb < 4; ++kb) {
    v8bf lo = *(const v8bf*)(abase + kb * 32);        // K = kb*32 + h8 + 0..7
    v8bf hi = *(const v8bf*)(abase + kb * 32 + 16);   // K = kb*32 + h8 + 16..23
    v16bf f;
#pragma unroll
    for (int i = 0; i < 8; ++i) { f[i] = lo[i]; f[i + 8] = hi[i]; }
    afr[kb] = f;
  }

  const __bf16* bbase = Wp + lane * 16;   // all fragments at imm offsets
#pragma unroll
  for (int nb = 0; nb < NBLK; ++nb) {
    v8f acc = {};
#pragma unroll
    for (int kb = 0; kb < 4; ++kb) {
      const __bf16* bptr = bbase + (nb * 4 + kb) * 512;   // 1024B steps
      v8bf lo = *(const v8bf*)bptr;
      v8bf hi = *(const v8bf*)(bptr + 8);
      v16bf bfr;
#pragma unroll
      for (int i = 0; i < 8; ++i) { bfr[i] = lo[i]; bfr[i + 8] = hi[i]; }
      acc = __builtin_amdgcn_wmma_f32_16x16x32_bf16(
          false, afr[kb], false, bfr, (short)0, acc, false, false);
    }
    // D layout: VGPR i -> M = i + 8*half, N = r.
    float* cbase = C + (long)m0 * NCOLS + (nb << 4) + r;
    int mh = half * 8;
#pragma unroll
    for (int i = 0; i < 8; ++i)
      cbase[(long)(mh + i) * NCOLS] = acc[i];
  }
}

// ----------------------------------------------------------- accumulator init
// agg[n][d] = xw[n][d] * dinv[n]^2 + b[d]   (self-loop message + bias).
// Folding this in up front removes the zero pass and the post-scatter add.
template <int SHIFT>   // SHIFT = log2(dim): 7 for 128, 6 for 64
__global__ void init_acc_kernel(float* __restrict__ agg,
                                const float* __restrict__ xw,
                                const float* __restrict__ dinv,
                                const float* __restrict__ b, long total) {
  long t = (long)blockIdx.x * blockDim.x + threadIdx.x;
  if (t >= total) return;
  int d = (int)(t & ((1 << SHIFT) - 1));
  int n = (int)(t >> SHIFT);
  float di = dinv[n];
  agg[t] = xw[t] * di * di + b[d];
}

// --------------------------------------------------------------- scatter
// One wave per edge: gather xw[src] (float4/float2 per lane), scale by
// dinv[src]*dinv[dst], atomic-accumulate into agg[dst].
template <int DIM>
__global__ void __launch_bounds__(256)
scatter_kernel(const int* __restrict__ src, const int* __restrict__ dst,
               const float* __restrict__ dinv, const float* __restrict__ xw,
               float* __restrict__ agg, int nE) {
  int lane = threadIdx.x & 31;
  long e = ((long)blockIdx.x * blockDim.x + threadIdx.x) >> 5;
  if (e >= nE) return;
  int s = src[e], d = dst[e];
  float w = dinv[s] * dinv[d];
  if (DIM == 128) {
    const float4 v = *(const float4*)(xw + (long)s * 128 + lane * 4);
    float* o = agg + (long)d * 128 + lane * 4;
    atomic_add_f32(o + 0, v.x * w);
    atomic_add_f32(o + 1, v.y * w);
    atomic_add_f32(o + 2, v.z * w);
    atomic_add_f32(o + 3, v.w * w);
  } else {  // DIM == 64
    const float2 v = *(const float2*)(xw + (long)s * 64 + lane * 2);
    float* o = agg + (long)d * 64 + lane * 2;
    atomic_add_f32(o + 0, v.x * w);
    atomic_add_f32(o + 1, v.y * w);
  }
}

// h = bf16(relu(agg)) for the second WMMA GEMM.
__global__ void relu_cvt_kernel(const float* __restrict__ agg,
                                __bf16* __restrict__ h, long total) {
  long t = (long)blockIdx.x * blockDim.x + threadIdx.x;
  if (t >= total) return;
  float v = agg[t];
  h[t] = (__bf16)(v > 0.f ? v : 0.f);
}

// ---------------------------------------------------------------- launch
extern "C" void kernel_launch(void* const* d_in, const int* in_sizes, int n_in,
                              void* d_out, int out_size, void* d_ws, size_t ws_size,
                              hipStream_t stream) {
  (void)n_in; (void)out_size; (void)ws_size;
  const float* x  = (const float*)d_in[0];
  const int*   ei = (const int*)d_in[1];
  const float* W1 = (const float*)d_in[2];
  const float* b1 = (const float*)d_in[3];
  const float* W2 = (const float*)d_in[4];
  const float* b2 = (const float*)d_in[5];
  float* out = (float*)d_out;

  const int IN  = 128;
  const int N   = in_sizes[0] / IN;     // 50000 (multiple of 16)
  const int E   = in_sizes[1] / 2;      // 1.6M
  const int HID = in_sizes[3];          // 128
  const int OUT = in_sizes[5];          // 64
  const int* src = ei;
  const int* dst = ei + E;

  // Workspace carving (256B aligned); xbf is reused as h_bf16 after GEMM1.
  char* p = (char*)d_ws;
  auto carve = [&](size_t bytes) -> char* {
    char* r = p; p += (bytes + 255) & ~(size_t)255; return r;
  };
  uint32_t* cnt  = (uint32_t*)carve((size_t)N * 4);
  float*    dinv = (float*)   carve((size_t)N * 4);
  __bf16*   xbf  = (__bf16*)  carve((size_t)N * IN  * 2);
  float*    xw1  = (float*)   carve((size_t)N * HID * 4);
  float*    agg1 = (float*)   carve((size_t)N * HID * 4);
  float*    xw2  = (float*)   carve((size_t)N * OUT * 4);
  __bf16*   W1p  = (__bf16*)  carve((size_t)IN  * HID * 2);
  __bf16*   W2p  = (__bf16*)  carve((size_t)HID * OUT * 2);

  const int B = 256;
  const long nx = (long)N * IN;     // N*128
  const long no = (long)N * OUT;    // N*64
  const int edgeBlocks = (int)(((long)E * 32 + B - 1) / B);
  const int gemmBlocks = ((N / 16) + 7) / 8;   // 8 waves/block, 16 rows/wave

  // Degrees & normalization
  zero_kernel<<<(N + B - 1) / B, B, 0, stream>>>(cnt, N);
  deg_kernel<<<(E + B - 1) / B, B, 0, stream>>>(dst, E, cnt);
  dinv_kernel<<<(N + B - 1) / B, B, 0, stream>>>(cnt, dinv, N);

  // Precision convert + weight swizzle
  cvt_bf16_kernel<<<(int)((nx + B - 1) / B), B, 0, stream>>>(x, xbf, nx);
  prep_w_kernel<<<(IN * HID + B - 1) / B, B, 0, stream>>>(W1, W1p, HID, IN * HID);
  prep_w_kernel<<<(HID * OUT + B - 1) / B, B, 0, stream>>>(W2, W2p, OUT, HID * OUT);

  // Layer 1: GEMM -> init(self-loop+bias) -> scatter -> relu+bf16 (into xbf)
  gemm_bf16_kernel<8><<<gemmBlocks, B, 0, stream>>>(xbf, W1p, xw1, N);
  init_acc_kernel<7><<<(int)((nx + B - 1) / B), B, 0, stream>>>(agg1, xw1, dinv, b1, nx);
  scatter_kernel<128><<<edgeBlocks, B, 0, stream>>>(src, dst, dinv, xw1, agg1, E);
  relu_cvt_kernel<<<(int)((nx + B - 1) / B), B, 0, stream>>>(agg1, xbf, nx);

  // Layer 2: GEMM -> init d_out(self-loop+bias) -> scatter into d_out (done)
  gemm_bf16_kernel<4><<<gemmBlocks, B, 0, stream>>>(xbf, W2p, xw2, N);
  init_acc_kernel<6><<<(int)((no + B - 1) / B), B, 0, stream>>>(out, xw2, dinv, b2, no);
  scatter_kernel<64><<<edgeBlocks, B, 0, stream>>>(src, dst, dinv, xw2, out, E);
}
